// SelfAttention_25804163514425
// MI455X (gfx1250) — matrix operand order repaired and checked by reading, not compiled, not run on hardware
//
#include <hip/hip_runtime.h>
#include <hip/hip_bf16.h>

#define S_LEN 2048
#define BATCH 2
#define EMB   1024
#define NH    16
#define HD    64
#define NHEADS (BATCH * NH)      // 32
#define M_ROWS (S_LEN * BATCH)   // 4096
#define N_QKV  (3 * EMB)         // 3072

typedef __attribute__((ext_vector_type(16))) __bf16 v16bf;
typedef __attribute__((ext_vector_type(8)))  __bf16 v8bf;
typedef __attribute__((ext_vector_type(4)))  __bf16 v4bf;
typedef __attribute__((ext_vector_type(8)))  float  v8f;

#define WMMA_BF16(a, b, c) \
  __builtin_amdgcn_wmma_f32_16x16x32_bf16(false, (a), false, (b), (short)0, (c), false, false)

__device__ __forceinline__ __bf16 f2bf(float f) {
  unsigned u = __builtin_bit_cast(unsigned, f);
  u += 0x7FFFu + ((u >> 16) & 1u);                 // round-to-nearest-even
  unsigned short s = (unsigned short)(u >> 16);
  return __builtin_bit_cast(__bf16, s);
}

// A fragment (16x32, MxK), row-major source with leading dim `ld`.
__device__ __forceinline__ v16bf load_frag_a(const __bf16* base, int ld, int mBase, int kBase, int lane) {
  const int row = mBase + (lane & 15);
  const int kh  = (lane >> 4) * 8;
  const __bf16* p = base + (size_t)row * ld + kBase + kh;
  v8bf lo = *(const v8bf*)(p);
  v8bf hi = *(const v8bf*)(p + 16);
  v16bf r;
#pragma unroll
  for (int i = 0; i < 8; ++i) { r[i] = lo[i]; r[i + 8] = hi[i]; }
  return r;
}

// B fragment (32x16, KxN), B[k][n] = src[nBase+n][kBase+k] (src row-major, rows are N).
__device__ __forceinline__ v16bf load_frag_b(const __bf16* base, int ld, int nBase, int kBase, int lane) {
  const int row = nBase + (lane & 15);
  const int kh  = (lane >> 4) * 8;
  const __bf16* p = base + (size_t)row * ld + kBase + kh;
  v8bf lo = *(const v8bf*)(p);
  v8bf hi = *(const v8bf*)(p + 16);
  v16bf r;
#pragma unroll
  for (int i = 0; i < 8; ++i) { r[i] = lo[i]; r[i + 8] = hi[i]; }
  return r;
}

// A fragment from f32 probabilities in LDS, converted to bf16 on the fly.
__device__ __forceinline__ v16bf load_frag_a_lds(const float* sS, int ld, int kBase, int lane) {
  const int row = lane & 15;
  const int kh  = (lane >> 4) * 8;
  const float* p = sS + row * ld + kBase + kh;
  float4 v0 = *(const float4*)(p);
  float4 v1 = *(const float4*)(p + 4);
  float4 v2 = *(const float4*)(p + 16);
  float4 v3 = *(const float4*)(p + 20);
  v16bf r;
  r[0]  = f2bf(v0.x); r[1]  = f2bf(v0.y); r[2]  = f2bf(v0.z); r[3]  = f2bf(v0.w);
  r[4]  = f2bf(v1.x); r[5]  = f2bf(v1.y); r[6]  = f2bf(v1.z); r[7]  = f2bf(v1.w);
  r[8]  = f2bf(v2.x); r[9]  = f2bf(v2.y); r[10] = f2bf(v2.z); r[11] = f2bf(v2.w);
  r[12] = f2bf(v3.x); r[13] = f2bf(v3.y); r[14] = f2bf(v3.z); r[15] = f2bf(v3.w);
  return r;
}

// ---- async global->LDS staging of a 64(N) x 64(K) bf16 panel (8 KiB), 256 threads ----
// Each thread issues one GLOBAL_LOAD_ASYNC_TO_LDS_B128 per 16B chunk (2 chunks/thread),
// tracked by ASYNCcnt.
__device__ __forceinline__ void stage_panel_async(const __bf16* W, int ldw, int nBase, int k0,
                                                  __bf16* tile, int tid) {
#pragma unroll
  for (int t = 0; t < 2; ++t) {
    const int c   = tid + t * 256;        // 0..511
    const int row = c >> 3;               // 64 rows
    const int col = (c & 7) * 8;          // 8 x 16B chunks per 128B row
    const __bf16* g = W + (size_t)(nBase + row) * ldw + k0 + col;
    const unsigned ldsa = (unsigned)(size_t)(tile + row * 64 + col);
    asm volatile("global_load_async_to_lds_b128 %0, %1, off"
                 :: "v"(ldsa), "v"((unsigned long long)(size_t)g)
                 : "memory");
  }
}

__device__ __forceinline__ void wait_async_all() {
  asm volatile("s_wait_asynccnt 0x0" ::: "memory");
}

// Shared GEMM mainloop: block = 8 waves; block tile M=128 x N=64; K chunk = 64,
// double-buffered LDS B panel fed by async copies; A streamed from global per wave.
__device__ __forceinline__ void gemm_mainloop(const __bf16* __restrict__ A, int lda,
                                              const __bf16* __restrict__ W, int Kdim,
                                              int mBase, int nBase,
                                              __bf16* sB, int tid, int lane, v8f acc[4]) {
  stage_panel_async(W, lda, nBase, 0, sB, tid);
  wait_async_all();
  __syncthreads();
  int cur = 0;
  for (int k0 = 0; k0 < Kdim; k0 += 64) {
    if (k0 + 64 < Kdim)
      stage_panel_async(W, lda, nBase, k0 + 64, sB + (cur ^ 1) * (64 * 64), tid);
    if (k0 + 64 < Kdim)
      __builtin_prefetch((const void*)(A + (size_t)(mBase + (lane & 15)) * lda + k0 + 64), 0, 3);
    v16bf a0 = load_frag_a(A, lda, mBase, k0, lane);
    v16bf a1 = load_frag_a(A, lda, mBase, k0 + 32, lane);
    const __bf16* tile = sB + cur * (64 * 64);
#pragma unroll
    for (int j = 0; j < 4; ++j) {
      acc[j] = WMMA_BF16(a0, load_frag_b(tile, 64, j * 16, 0,  lane), acc[j]);
      acc[j] = WMMA_BF16(a1, load_frag_b(tile, 64, j * 16, 32, lane), acc[j]);
    }
    wait_async_all();      // staged panel for next chunk is resident
    __syncthreads();       // all waves done reading current panel
    cur ^= 1;
  }
}

// ---------------- f32 -> bf16 conversion ----------------
__global__ void k_cvt(const float* __restrict__ src, __bf16* __restrict__ dst, int n) {
  int i = (blockIdx.x * blockDim.x + threadIdx.x) * 4;
  if (i < n) {
    float4 v = *(const float4*)(src + i);
    v4bf o;
    o[0] = f2bf(v.x); o[1] = f2bf(v.y); o[2] = f2bf(v.z); o[3] = f2bf(v.w);
    *(v4bf*)(dst + i) = o;
  }
}

// ---------------- QKV projection: [4096,3072] = xb @ Winb^T + bin, scatter to head layouts ----
__global__ __launch_bounds__(256) void k_qkv(const __bf16* __restrict__ X,
                                             const __bf16* __restrict__ W,
                                             const float* __restrict__ binv,
                                             __bf16* __restrict__ Q,
                                             __bf16* __restrict__ K,
                                             __bf16* __restrict__ Vt) {
  __shared__ __bf16 sB[2 * 64 * 64];           // 16 KiB double-buffered B panel
  const int tid  = threadIdx.x;
  const int lane = tid & 31;
  const int wave = tid >> 5;
  const int mBase = blockIdx.y * 128 + wave * 16;
  const int nBase = blockIdx.x * 64;
  v8f acc[4] = {};
  gemm_mainloop(X, EMB, W, EMB, mBase, nBase, sB, tid, lane, acc);

  const int rhalf = (lane >> 4) * 8;
#pragma unroll
  for (int j = 0; j < 4; ++j) {
    const int col   = nBase + j * 16 + (lane & 15);
    const int which = col >> 10;      // 0=q, 1=k, 2=v
    const int e     = col & 1023;
    const int h     = e >> 6;
    const int d     = e & 63;
    const float bv  = binv[col];
#pragma unroll
    for (int i = 0; i < 8; ++i) {
      const int row = mBase + i + rhalf;   // row = s*BATCH + b
      const int s   = row >> 1;
      const int b_  = row & 1;
      const int n   = b_ * NH + h;
      float v = acc[j][i] + bv;
      if (which == 0)      Q[((size_t)(n * S_LEN + s)) * HD + d] = f2bf(v * 0.125f); // fold 1/sqrt(64)
      else if (which == 1) K[((size_t)(n * S_LEN + s)) * HD + d] = f2bf(v);
      else                 Vt[((size_t)(n * HD + d)) * S_LEN + s] = f2bf(v);          // V transposed
    }
  }
}

// ---------------- attention: one block per (head, 16-query tile) ----------------
__global__ __launch_bounds__(256) void k_attn(const __bf16* __restrict__ Q,
                                              const __bf16* __restrict__ K,
                                              const __bf16* __restrict__ Vt,
                                              const unsigned char* __restrict__ mask,
                                              __bf16* __restrict__ O) {
  extern __shared__ float sS[];            // [16][2048] scores, then red[16*16]
  float* red = sS + 16 * S_LEN;
  const int tid  = threadIdx.x;
  const int lane = tid & 31;
  const int wave = tid >> 5;
  const int head = blockIdx.y;
  const int qBase = blockIdx.x * 16;
  const __bf16* Qh = Q  + (size_t)head * S_LEN * HD;
  const __bf16* Kh = K  + (size_t)head * S_LEN * HD;
  const __bf16* Vh = Vt + (size_t)head * HD * S_LEN;
  const int rhalf = (lane >> 4) * 8;

  // Phase 1: scores[16][2048] = Qtile @ K^T  (scale pre-folded into Q)
  v16bf a0 = load_frag_a(Qh, HD, qBase, 0, lane);
  v16bf a1 = load_frag_a(Qh, HD, qBase, 32, lane);
  for (int it = 0; it < 16; ++it) {
    const int tBase = (wave * 16 + it) * 16;
    v8f c = {};
    c = WMMA_BF16(a0, load_frag_b(Kh, HD, tBase, 0, lane), c);
    c = WMMA_BF16(a1, load_frag_b(Kh, HD, tBase, 32, lane), c);
    const int n = tBase + (lane & 15);
#pragma unroll
    for (int i = 0; i < 8; ++i) sS[(i + rhalf) * S_LEN + n] = c[i];
  }
  __syncthreads();

  // Phase 2: masked softmax over each row (16 threads per row, 128 cols each)
  const int row = tid & 15;
  const int seg = tid >> 4;
  const unsigned char* mrow = mask + (size_t)(head >> 4) * S_LEN;   // b = head / NH
  float* r0 = sS + row * S_LEN + seg * 128;
  float lmax = -1e30f;
  for (int c = 0; c < 128; ++c) {
    float v = r0[c];
    if (mrow[seg * 128 + c]) { v = -1e30f; r0[c] = v; }
    lmax = fmaxf(lmax, v);
  }
  red[row * 16 + seg] = lmax;
  __syncthreads();
  float m = -1e30f;
  for (int s2 = 0; s2 < 16; ++s2) m = fmaxf(m, red[row * 16 + s2]);
  __syncthreads();
  float lsum = 0.f;
  for (int c = 0; c < 128; ++c) {
    float e = __expf(r0[c] - m);
    r0[c] = e;
    lsum += e;
  }
  red[row * 16 + seg] = lsum;
  __syncthreads();
  float sum = 0.f;
  for (int s2 = 0; s2 < 16; ++s2) sum += red[row * 16 + s2];
  const float inv = 1.f / sum;
  for (int c = 0; c < 128; ++c) r0[c] *= inv;
  __syncthreads();

  // Phase 3: O[16][64] = P @ V. All 8 waves: wave = kh*4 + j; split-K then LDS reduce.
  const int j  = wave & 3;           // 16-wide d slice
  const int kh = wave >> 2;          // K half: 0 -> [0,1024), 1 -> [1024,2048)
  v8f o = {};
  for (int k0 = kh * 1024; k0 < kh * 1024 + 1024; k0 += 32) {
    v16bf a = load_frag_a_lds(sS, S_LEN, k0, lane);
    v16bf b = load_frag_b(Vh, S_LEN, j * 16, k0, lane);
    o = WMMA_BF16(a, b, o);
  }
  __syncthreads();                   // everyone done reading sS; reuse it as scratch
  float* scratch = sS;               // 4 waves * 32 lanes * 8 floats = 4 KiB
  if (kh == 1) {
#pragma unroll
    for (int i = 0; i < 8; ++i) scratch[(j * 32 + lane) * 8 + i] = o[i];
  }
  __syncthreads();
  if (kh == 0) {
    const int d  = j * 16 + (lane & 15);
    const int b_ = head >> 4;
    const int h  = head & 15;
#pragma unroll
    for (int i = 0; i < 8; ++i) {
      const float val = o[i] + scratch[(j * 32 + lane) * 8 + i];
      const int s = qBase + i + rhalf;
      O[(size_t)(s * BATCH + b_) * EMB + h * HD + d] = f2bf(val);
    }
  }
}

// ---------------- output projection: out[4096,1024] = Ob @ Woutb^T + bout ----------------
__global__ __launch_bounds__(256) void k_out(const __bf16* __restrict__ A,
                                             const __bf16* __restrict__ W,
                                             const float* __restrict__ boutv,
                                             float* __restrict__ out) {
  __shared__ __bf16 sB[2 * 64 * 64];
  const int tid  = threadIdx.x;
  const int lane = tid & 31;
  const int wave = tid >> 5;
  const int mBase = blockIdx.y * 128 + wave * 16;
  const int nBase = blockIdx.x * 64;
  v8f acc[4] = {};
  gemm_mainloop(A, EMB, W, EMB, mBase, nBase, sB, tid, lane, acc);

  const int rhalf = (lane >> 4) * 8;
#pragma unroll
  for (int j = 0; j < 4; ++j) {
    const int col  = nBase + j * 16 + (lane & 15);
    const float bv = boutv[col];
#pragma unroll
    for (int i = 0; i < 8; ++i) {
      const int row = mBase + i + rhalf;
      out[(size_t)row * EMB + col] = acc[j][i] + bv;
    }
  }
}

extern "C" void kernel_launch(void* const* d_in, const int* in_sizes, int n_in,
                              void* d_out, int out_size, void* d_ws, size_t ws_size,
                              hipStream_t stream) {
  (void)in_sizes; (void)n_in; (void)out_size; (void)ws_size;
  const float*         x    = (const float*)d_in[0];
  const unsigned char* mask = (const unsigned char*)d_in[1];
  const float*         Win  = (const float*)d_in[2];
  const float*         bin  = (const float*)d_in[3];
  const float*         Wout = (const float*)d_in[4];
  const float*         bout = (const float*)d_in[5];
  float* out = (float*)d_out;

  char* ws = (char*)d_ws;                      // ~48 MiB used
  __bf16* xb    = (__bf16*)(ws);                       //  8 MiB  [4096,1024]
  __bf16* Winb  = (__bf16*)(ws + ((size_t)8  << 20));  //  6 MiB  [3072,1024]
  __bf16* Woutb = (__bf16*)(ws + ((size_t)14 << 20));  //  2 MiB  [1024,1024]
  __bf16* Qb    = (__bf16*)(ws + ((size_t)16 << 20));  //  8 MiB  [32][2048][64]
  __bf16* Kb    = (__bf16*)(ws + ((size_t)24 << 20));  //  8 MiB  [32][2048][64]
  __bf16* Vtb   = (__bf16*)(ws + ((size_t)32 << 20));  //  8 MiB  [32][64][2048]
  __bf16* Ob    = (__bf16*)(ws + ((size_t)40 << 20));  //  8 MiB  [4096,1024]

  const int nx  = M_ROWS * EMB;
  const int nwi = N_QKV * EMB;
  const int nwo = EMB * EMB;
  k_cvt<<<nx  / 1024, 256, 0, stream>>>(x,    xb,    nx);
  k_cvt<<<nwi / 1024, 256, 0, stream>>>(Win,  Winb,  nwi);
  k_cvt<<<nwo / 1024, 256, 0, stream>>>(Wout, Woutb, nwo);

  // block tile: M=128 (8 waves x 16) x N=64
  k_qkv<<<dim3(N_QKV / 64, M_ROWS / 128), 256, 0, stream>>>(xb, Winb, bin, Qb, Kb, Vtb);

  const int smem = (16 * S_LEN + 16 * 16) * (int)sizeof(float);   // ~132 KB
  hipFuncSetAttribute((const void*)k_attn, hipFuncAttributeMaxDynamicSharedMemorySize, smem);
  k_attn<<<dim3(S_LEN / 16, NHEADS), 256, smem, stream>>>(Qb, Kb, Vtb, mask, Ob);

  k_out<<<dim3(EMB / 64, M_ROWS / 128), 256, 0, stream>>>(Ob, Woutb, bout, out);
}
